// MACE_29850022707543
// MI455X (gfx1250) — matrix operand order repaired and checked by reading, not compiled, not run on hardware
//
#include <hip/hip_runtime.h>
#include <hip/hip_bf16.h>
#include <math.h>

// MACE forward for MI455X (gfx1250, wave32).
// N=10000 nodes, E=160000 edges, F=64, S=10, NR=8.
#define NN 10000
#define EE 160000
#define FF 64
#define SS 10
#define SC_EPS (1.0f / 17.0f)   // EPS_C^2 = 1/(1+16)

typedef __attribute__((ext_vector_type(16))) _Float16 v16h;
typedef __attribute__((ext_vector_type(8)))  float    v8f;
typedef __attribute__((ext_vector_type(2)))  float    v2f;
typedef __attribute__((ext_vector_type(4)))  unsigned int u32x4;
typedef __attribute__((ext_vector_type(8)))  int      i32x8;
typedef __attribute__((ext_vector_type(4)))  int      i32x4;
typedef __attribute__((address_space(3)))    _Float16 lds_f16_t;

__device__ __forceinline__ float silu_f(float x) { return x / (1.0f + __expf(-x)); }

// ---------------------------------------------------------------------------
// zero workspace region
__global__ void mace_zero_f32(float* __restrict__ p, int n) {
  int i = blockIdx.x * blockDim.x + threadIdx.x;
  if (i < n) p[i] = 0.0f;
}

// ---------------------------------------------------------------------------
// convert r2 (32 x 448 f32) into WMMA B-fragment order (f16):
// tile t (28 tiles of 16 cols), lane L (32), half i (16):
//   K = i + (L>=16 ? 16 : 0), col = t*16 + (L&15)
__global__ void mace_r2_to_frag(const float* __restrict__ r2, _Float16* __restrict__ out) {
  int idx = blockIdx.x * blockDim.x + threadIdx.x;     // 28*32*16 = 14336
  if (idx >= 28 * 32 * 16) return;
  int t = idx >> 9;
  int L = (idx >> 4) & 31;
  int i = idx & 15;
  int K = i + ((L & 16) ? 16 : 0);
  int col = t * 16 + (L & 15);
  out[idx] = (_Float16)r2[K * 448 + col];
}

// ---------------------------------------------------------------------------
// s = embed[species]
__global__ void mace_node_init(const int* __restrict__ spec, const float* __restrict__ embed,
                               float* __restrict__ s) {
  int i = blockIdx.x * blockDim.x + threadIdx.x;
  if (i < NN * FF) s[i] = embed[spec[i >> 6] * FF + (i & 63)];
}

// ---------------------------------------------------------------------------
// Fused edge kernel: geometry + radial basis + (silu(rbf@r1))@r2 via WMMA +
// message construction + atomic scatter into Bacc[N][64][16].
// One wave handles 16 edges. Block = 128 threads = 4 waves = 64 edges.
// The pre-swizzled r2 B-fragment table (28 KB) is DMA'd into LDS once per
// block by the Tensor Data Mover so per-tile B reads ride the LDS path and
// leave L2 ports free for the atomic scatter stream (the roofline limiter).
template <bool HAS_V>
__global__ __launch_bounds__(128) void mace_edge_msg(
    const float* __restrict__ vectors,   // (E,3)
    const int*   __restrict__ senders,
    const int*   __restrict__ receivers,
    const float* __restrict__ r1,        // (8,32)
    const _Float16* __restrict__ r2f,    // B-fragment layout, 28*32*16 halves
    const float* __restrict__ s_in,      // (N,64)
    const float* __restrict__ v_in,      // (N,64,3)
    float* __restrict__ Bacc)            // (N,64,16)
{
  __shared__ __align__(32) _Float16 r2lds[28 * 32 * 16];  // 28 KB TDM stage
  __shared__ _Float16 Wl[4][16][448];                     // per-wave W tile (56 KB)

  const int lane = threadIdx.x & 31;
  const int wave = threadIdx.x >> 5;
  const int hi   = lane >> 4;            // wave32 half
  const int el   = lane & 15;            // local edge row
  const int e    = (blockIdx.x * 4 + wave) * 16 + el;  // E divisible by 64

#if __has_builtin(__builtin_amdgcn_tensor_load_to_lds)
  if (wave == 0) {
    // D# for a 1-D copy: 7168 dwords (28 KB), data_size=4B, tile == tensor.
    unsigned ldsoff = (unsigned)(unsigned long long)(lds_f16_t*)&r2lds[0];
    unsigned long long ga = (unsigned long long)(const void*)r2f;
    u32x4 g0;
    g0[0] = 1u;                                            // count=1 (valid)
    g0[1] = ldsoff;                                        // lds_addr
    g0[2] = (unsigned)(ga & 0xffffffffu);                  // global_addr[31:0]
    g0[3] = (unsigned)((ga >> 32) & 0x01ffffffu) | (2u << 30); // addr[56:32] | type=2
    i32x8 g1;
    g1[0] = (int)(2u << 16);          // workgroup_mask=0, data_size=4B
    g1[1] = (int)(0x1C00u << 16);     // tensor_dim0 = 7168 (low 16 bits)
    g1[2] = (int)(1u << 16);          // tensor_dim0 hi=0, tensor_dim1 = 1
    g1[3] = (int)(0x1C00u << 16);     // tensor_dim1 hi=0, tile_dim0 = 7168
    g1[4] = 1;                        // tile_dim1 = 1, tile_dim2 = 0
    g1[5] = 0x1C00;                   // tensor_dim0_stride = 7168
    g1[6] = 0;
    g1[7] = 0;
    i32x4 g2 = {0, 0, 0, 0};
    i32x4 g3 = {0, 0, 0, 0};
    i32x8 g4 = {0, 0, 0, 0, 0, 0, 0, 0};
    __builtin_amdgcn_tensor_load_to_lds(g0, g1, g2, g3, g4, 0);
    __builtin_amdgcn_s_wait_tensorcnt(0);
  }
#else
  for (int i = threadIdx.x; i < 28 * 32 * 16; i += 128) r2lds[i] = r2f[i];
#endif
  __syncthreads();

  // ---- geometry (both halves redundantly compute their edge 'el') ----
  float vx = vectors[e * 3 + 0], vy = vectors[e * 3 + 1], vz = vectors[e * 3 + 2];
  float rn = sqrtf(vx * vx + vy * vy + vz * vz + 1e-12f);
  float inv = 1.0f / rn;
  float ux = vx * inv, uy = vy * inv, uz = vz * inv;
  const float c3  = 1.7320508075688772f;
  const float s58 = 0.7905694150420949f;
  const float s15 = 3.872983346207417f;
  const float s38 = 0.6123724356957945f;
  float y[16];
  y[0] = 1.0f;
  y[1] = ux; y[2] = uy; y[3] = uz;
  y[4] = c3 * ux * uy;
  y[5] = c3 * uy * uz;
  y[6] = 0.5f * (3.f * uz * uz - 1.f);
  y[7] = c3 * ux * uz;
  y[8] = 0.5f * c3 * (ux * ux - uy * uy);
  y[9]  = s58 * uy * (3.f * ux * ux - uy * uy);
  y[10] = s15 * ux * uy * uz;
  y[11] = s38 * uy * (5.f * uz * uz - 1.f);
  y[12] = 0.5f * uz * (5.f * uz * uz - 3.f);
  y[13] = s38 * ux * (5.f * uz * uz - 1.f);
  y[14] = 0.5f * s15 * uz * (ux * ux - uy * uy);
  y[15] = s58 * ux * (ux * ux - 3.f * uy * uy);

  // ---- radial basis (p=6 envelope, 8 sine bessels) ----
  float xcl = fminf(rn, 1.0f);
  float x3 = xcl * xcl * xcl;
  float x6 = x3 * x3;
  float env = 1.f - 28.f * x6 + 48.f * x6 * xcl - 21.f * x6 * xcl * xcl;
  float pref = 1.4142135623730951f * env / (rn + 1e-9f);
  float rbf[8];
#pragma unroll
  for (int n = 0; n < 8; ++n)
    rbf[n] = pref * __sinf((float)(n + 1) * 3.14159265358979323846f * xcl);

  // ---- h = silu(rbf @ r1), built directly in WMMA A-fragment lane layout ----
  // contiguous K order chosen here (same as r2 frag): lanes 0-15 hold K {0..15}, lanes 16-31 hold K {16..31}
  v16h afrag;
#pragma unroll
  for (int i = 0; i < 16; ++i) {
    int K = i + (hi ? 16 : 0);
    float acc = 0.f;
#pragma unroll
    for (int j = 0; j < 8; ++j) acc += rbf[j] * r1[j * 32 + K];
    afrag[i] = (_Float16)silu_f(acc);
  }

  // ---- W = h @ r2 : 28 WMMA tiles of 16 edges x 16 cols, K=32 ----
#pragma unroll 4
  for (int t = 0; t < 28; ++t) {
    v16h bfrag = *(const v16h*)(&r2lds[(((t * 32) + lane) << 4)]);
    v8f cacc = {};
    cacc = __builtin_amdgcn_wmma_f32_16x16x32_f16(false, afrag, false, bfrag,
                                                  (short)0, cacc, false, false);
    // D layout: lane holds col N=lane&15, rows M = r + (hi?8:0)
#pragma unroll
    for (int r = 0; r < 8; ++r)
      Wl[wave][r + (hi ? 8 : 0)][t * 16 + el] = (_Float16)cacc[r];
  }
  __syncthreads();

  // ---- message phase: lane pair (l, l+16) splits 64 features of edge el ----
  const int snd = senders[e];
  const int rcv = receivers[e];
  const int f0  = hi ? 32 : 0;
  float* Bp = Bacc + (size_t)rcv * (FF * 16);
  const float* sp = s_in + (size_t)snd * FF;
  const float* vp = v_in + (size_t)snd * FF * 3;

  for (int fi = 0; fi < 32; ++fi) {
    int f = f0 + fi;
    float ss = sp[f];
    float w0 = (float)Wl[wave][el][0 * 64 + f];
    float w1 = (float)Wl[wave][el][1 * 64 + f];
    float w2 = (float)Wl[wave][el][2 * 64 + f];
    float w3 = (float)Wl[wave][el][3 * 64 + f];
    float w4 = (float)Wl[wave][el][4 * 64 + f];
    float w5 = (float)Wl[wave][el][5 * 64 + f];
    float w6 = (float)Wl[wave][el][6 * 64 + f];

    float m[16];
    float m0 = w0 * ss;
    float m1x = w2 * ss * y[1], m1y = w2 * ss * y[2], m1z = w2 * ss * y[3];
    if (HAS_V) {
      float a0 = vp[f * 3 + 0], a1 = vp[f * 3 + 1], a2 = vp[f * 3 + 2];
      m0 += w1 * (a0 * y[1] + a1 * y[2] + a2 * y[3]);
      float cx = a1 * y[3] - a2 * y[2];
      float cy = a2 * y[1] - a0 * y[3];
      float cz = a0 * y[2] - a1 * y[1];
      m1x += w3 * a0 + w4 * cx;
      m1y += w3 * a1 + w4 * cy;
      m1z += w3 * a2 + w4 * cz;
    }
    m[0] = m0; m[1] = m1x; m[2] = m1y; m[3] = m1z;
#pragma unroll
    for (int j = 0; j < 5; ++j) m[4 + j] = w5 * ss * y[4 + j];
#pragma unroll
    for (int j = 0; j < 7; ++j) m[9 + j] = w6 * ss * y[9 + j];

    float* dst = Bp + f * 16;
#pragma unroll
    for (int k = 0; k < 16; ++k)
      (void)__hip_atomic_fetch_add(dst + k, SC_EPS * m[k],
                                   __ATOMIC_RELAXED, __HIP_MEMORY_SCOPE_AGENT);
  }
}

// ---------------------------------------------------------------------------
// A[n][g][k] = sum_f Bacc[n][f][k] * mix[slice(k)][f][g], full-f32 WMMA
// (V_WMMA_F32_16X16X4_F32, K-loop of 16 steps). One wave per (ntile, k, gtile).
__global__ __launch_bounds__(256) void mace_mix_wmma(
    const float* __restrict__ Bacc,  // (N,64,16)
    const float* __restrict__ mix,   // (4,64,64)
    float* __restrict__ A)           // (N,64,16)
{
  const int lane = threadIdx.x & 31;
  const int wid  = blockIdx.x * 8 + (threadIdx.x >> 5);   // 40000 waves
  const int gt   = wid & 3;
  const int kk   = (wid >> 2) & 15;
  const int nt   = wid >> 6;                               // 0..624
  const int hi   = lane >> 4;
  const int lm   = lane & 15;
  const int slice = (kk == 0) ? 0 : (kk < 4) ? 1 : (kk < 9) ? 2 : 3;
  const float* Wm = mix + slice * 64 * 64;
  const float* Bp = Bacc + ((size_t)(nt * 16 + lm) * 64) * 16 + kk;

  v8f cacc = {};
#pragma unroll 4
  for (int fs = 0; fs < 64; fs += 4) {
    int fa = fs + (hi ? 2 : 0);
    v2f av, bv;
    av.x = Bp[(size_t)fa * 16];
    av.y = Bp[(size_t)(fa + 1) * 16];
    bv.x = Wm[fa * 64 + gt * 16 + lm];
    bv.y = Wm[(fa + 1) * 64 + gt * 16 + lm];
    cacc = __builtin_amdgcn_wmma_f32_16x16x4_f32(false, av, false, bv,
                                                 (short)0, cacc, false, false);
  }
#pragma unroll
  for (int r = 0; r < 8; ++r) {
    int n = nt * 16 + r + (hi ? 8 : 0);
    int g = gt * 16 + lm;
    A[((size_t)n * 64 + g) * 16 + kk] = cacc[r];
  }
}

// ---------------------------------------------------------------------------
// Post-interaction-0 node kernel: species linear (sf_0), product basis (c_0),
// v_h (d_0), sc0 = s_h @ sc_s[spec].  4 nodes/block, 64 threads/node.
__global__ __launch_bounds__(256) void mace_node_post1(
    const float* __restrict__ A, const int* __restrict__ spec,
    const float* __restrict__ sf, const float* __restrict__ c0,
    const float* __restrict__ d0, const float* __restrict__ scs,
    float* __restrict__ s_out, float* __restrict__ v_out, float* __restrict__ sc0)
{
  __shared__ float Al[4][64][16];
  __shared__ float shl[4][64];
  const int ln = threadIdx.x >> 6;
  const int g  = threadIdx.x & 63;
  const int n  = blockIdx.x * 4 + ln;
  const int sp = spec[n];
#pragma unroll
  for (int k = 0; k < 16; ++k) Al[ln][g][k] = A[((size_t)n * 64 + g) * 16 + k];
  __syncthreads();

  const float* W0 = sf + (size_t)(0 * SS + sp) * 64 * 64;
  const float* W1 = sf + (size_t)(1 * SS + sp) * 64 * 64;
  const float* W2 = sf + (size_t)(2 * SS + sp) * 64 * 64;
  const float* W3 = sf + (size_t)(3 * SS + sp) * 64 * 64;
  float a0 = 0.f, a1[3] = {}, a2[5] = {}, a3[7] = {};
  for (int f = 0; f < 64; ++f) {
    float w0 = W0[f * 64 + g], w1 = W1[f * 64 + g];
    float w2 = W2[f * 64 + g], w3 = W3[f * 64 + g];
    const float* Af = Al[ln][f];
    a0 += Af[0] * w0;
#pragma unroll
    for (int k = 0; k < 3; ++k) a1[k] += Af[1 + k] * w1;
#pragma unroll
    for (int k = 0; k < 5; ++k) a2[k] += Af[4 + k] * w2;
#pragma unroll
    for (int k = 0; k < 7; ++k) a3[k] += Af[9 + k] * w3;
  }
  float i1 = 0.f, i2 = 0.f, i3 = 0.f;
#pragma unroll
  for (int k = 0; k < 3; ++k) i1 += a1[k] * a1[k];
#pragma unroll
  for (int k = 0; k < 5; ++k) i2 += a2[k] * a2[k];
#pragma unroll
  for (int k = 0; k < 7; ++k) i3 += a3[k] * a3[k];
  const float* cs = c0 + ((size_t)sp * 64 + g) * 6;
  float sh = cs[0] * a0 + cs[1] * a0 * a0 + cs[2] * a0 * a0 * a0
           + cs[3] * i1 + cs[4] * i2 + cs[5] * i3;
  const float* ds = d0 + ((size_t)sp * 64 + g) * 3;
  float fac = ds[0] + ds[1] * a0 + ds[2] * a0 * a0;
#pragma unroll
  for (int k = 0; k < 3; ++k) v_out[((size_t)n * 64 + g) * 3 + k] = a1[k] * fac;
  s_out[(size_t)n * 64 + g] = sh;
  shl[ln][g] = sh;
  __syncthreads();
  float acc = 0.f;
  for (int f = 0; f < 64; ++f) acc += shl[ln][f] * scs[((size_t)sp * 64 + f) * 64 + g];
  sc0[(size_t)n * 64 + g] = acc;
}

// ---------------------------------------------------------------------------
// Post-interaction-1: product basis (c_1) + sc0, then readout silu(s@rm1)@rm2.
__global__ __launch_bounds__(256) void mace_node_post2(
    const float* __restrict__ A, const int* __restrict__ spec,
    const float* __restrict__ c1, const float* __restrict__ sc0,
    const float* __restrict__ rm1, const float* __restrict__ rm2,
    float* __restrict__ out)
{
  __shared__ float sl[4][64];
  __shared__ float tl[4][16];
  const int ln = threadIdx.x >> 6;
  const int g  = threadIdx.x & 63;
  const int n  = blockIdx.x * 4 + ln;
  const int sp = spec[n];
  const float* Ap = A + ((size_t)n * 64 + g) * 16;
  float a0 = Ap[0];
  float i1 = 0.f, i2 = 0.f, i3 = 0.f;
#pragma unroll
  for (int k = 0; k < 3; ++k) { float v = Ap[1 + k]; i1 += v * v; }
#pragma unroll
  for (int k = 0; k < 5; ++k) { float v = Ap[4 + k]; i2 += v * v; }
#pragma unroll
  for (int k = 0; k < 7; ++k) { float v = Ap[9 + k]; i3 += v * v; }
  const float* cs = c1 + ((size_t)sp * 64 + g) * 6;
  float s = cs[0] * a0 + cs[1] * a0 * a0 + cs[2] * a0 * a0 * a0
          + cs[3] * i1 + cs[4] * i2 + cs[5] * i3 + sc0[(size_t)n * 64 + g];
  sl[ln][g] = s;
  __syncthreads();
  if (g < 16) {
    float t = 0.f;
    for (int f = 0; f < 64; ++f) t += sl[ln][f] * rm1[f * 16 + g];
    tl[ln][g] = silu_f(t);
  }
  __syncthreads();
  if (g == 0) {
    float o = 0.f;
#pragma unroll
    for (int j = 0; j < 16; ++j) o += tl[ln][j] * rm2[j];
    out[n] = o;
  }
}

// ---------------------------------------------------------------------------
extern "C" void kernel_launch(void* const* d_in, const int* in_sizes, int n_in,
                              void* d_out, int out_size, void* d_ws, size_t ws_size,
                              hipStream_t stream) {
  (void)in_sizes; (void)n_in; (void)out_size; (void)ws_size;
  const float* vectors  = (const float*)d_in[0];
  const int*   senders  = (const int*)d_in[1];
  const int*   receivers= (const int*)d_in[2];
  const int*   species  = (const int*)d_in[3];
  const float* embed    = (const float*)d_in[4];
  const float* r1_0     = (const float*)d_in[5];
  const float* r2_0     = (const float*)d_in[6];
  const float* mix_0    = (const float*)d_in[7];
  const float* sf_0     = (const float*)d_in[8];
  const float* c_0      = (const float*)d_in[9];
  const float* d_0      = (const float*)d_in[10];
  // d_in[11] = ro_0: result of s_h @ ro_0 is unused in the reference -> skipped
  const float* r1_1     = (const float*)d_in[12];
  const float* r2_1     = (const float*)d_in[13];
  const float* mix_1    = (const float*)d_in[14];
  const float* sc_s     = (const float*)d_in[15];
  const float* c_1      = (const float*)d_in[16];
  const float* rm1      = (const float*)d_in[17];
  const float* rm2      = (const float*)d_in[18];

  char* ws = (char*)d_ws;
  size_t off = 0;
  auto carve = [&](size_t bytes) -> void* {
    void* p = ws + off;
    off = (off + bytes + 255) & ~(size_t)255;
    return p;
  };
  float*    B    = (float*)carve((size_t)NN * 64 * 16 * 4);   // 40.96 MB
  float*    A    = (float*)carve((size_t)NN * 64 * 16 * 4);   // 40.96 MB
  float*    sbuf = (float*)carve((size_t)NN * 64 * 4);
  float*    vbuf = (float*)carve((size_t)NN * 64 * 3 * 4);
  float*    sc0  = (float*)carve((size_t)NN * 64 * 4);
  _Float16* r2h0 = (_Float16*)carve((size_t)28 * 32 * 16 * 2);
  _Float16* r2h1 = (_Float16*)carve((size_t)28 * 32 * 16 * 2);

  mace_r2_to_frag<<<56, 256, 0, stream>>>(r2_0, r2h0);
  mace_r2_to_frag<<<56, 256, 0, stream>>>(r2_1, r2h1);
  mace_node_init<<<(NN * FF + 255) / 256, 256, 0, stream>>>(species, embed, sbuf);

  // ---- interaction 0 (v = 0) ----
  mace_zero_f32<<<(NN * 64 * 16 + 255) / 256, 256, 0, stream>>>(B, NN * 64 * 16);
  mace_edge_msg<false><<<EE / 64, 128, 0, stream>>>(vectors, senders, receivers,
                                                    r1_0, r2h0, sbuf, nullptr, B);
  mace_mix_wmma<<<5000, 256, 0, stream>>>(B, mix_0, A);
  mace_node_post1<<<NN / 4, 256, 0, stream>>>(A, species, sf_0, c_0, d_0, sc_s,
                                              sbuf, vbuf, sc0);

  // ---- interaction 1 (with vectors v_h) ----
  mace_zero_f32<<<(NN * 64 * 16 + 255) / 256, 256, 0, stream>>>(B, NN * 64 * 16);
  mace_edge_msg<true><<<EE / 64, 128, 0, stream>>>(vectors, senders, receivers,
                                                   r1_1, r2h1, sbuf, vbuf, B);
  mace_mix_wmma<<<5000, 256, 0, stream>>>(B, mix_1, A);
  mace_node_post2<<<NN / 4, 256, 0, stream>>>(A, species, c_1, sc0, rm1, rm2,
                                              (float*)d_out);
}
